// BoltzAffinityHeadReplica_42133629174267
// MI455X (gfx1250) — matrix operand order repaired and checked by reading, not compiled, not run on hardware
//
#include <hip/hip_runtime.h>
#include <math.h>

#define LC   768
#define CP   128
#define CS   384
#define BB   64
#define HID  256
#define AH   128
#define EP   300000
#define INV_TEMP 0.25f
#define LN_EPS 1e-5f

typedef __attribute__((ext_vector_type(16))) __bf16 v16bf;
typedef __attribute__((ext_vector_type(8)))  __bf16 v8bf;
typedef __attribute__((ext_vector_type(8)))  float  v8f;
typedef int v4i_t __attribute__((vector_size(16)));

union AFrag { v16bf v; v8bf h[2]; };

#define HAS_ASYNC_LDS  __has_builtin(__builtin_amdgcn_global_load_async_to_lds_b128)

// ---------------------------------------------------------------- u,v projection
__global__ __launch_bounds__(256) void k_proj_uv(
    const float* __restrict__ s_proxy,
    const float* __restrict__ Wpu, const float* __restrict__ bpu,
    const float* __restrict__ Wpv, const float* __restrict__ bpv,
    float* __restrict__ u, float* __restrict__ v)
{
    __shared__ float srow[CS];
    const int row = blockIdx.x, t = threadIdx.x;
    for (int k = t; k < CS; k += 256) srow[k] = s_proxy[row * CS + k];
    __syncthreads();
    float au = bpu[t], av = bpv[t];
    #pragma unroll 4
    for (int k = 0; k < CS; ++k) {
        float s = srow[k];
        au += s * Wpu[t * CS + k];
        av += s * Wpv[t * CS + k];
    }
    u[row * HID + t] = au;
    v[row * HID + t] = av;
}

// ---------------------------------------------------------------- f32 -> bf16 weight copies
__global__ __launch_bounds__(256) void k_cvt(
    const float* __restrict__ Wbias, const float* __restrict__ Wdist,
    const float* __restrict__ Wa1,
    __bf16* __restrict__ WbB, __bf16* __restrict__ WdB, __bf16* __restrict__ Wa1B)
{
    int i = blockIdx.x * 256 + threadIdx.x;
    if (i < CP * 3 * HID) WbB[i]  = (__bf16)Wbias[i];
    if (i < CP * BB)      WdB[i]  = (__bf16)Wdist[i];
    if (i < AH * (2*CP))  Wa1B[i] = (__bf16)Wa1[i];
}

// ---------------------------------------------------------------- heavy per-pair kernel
__global__ __launch_bounds__(256) void k_pairs(
    const float* __restrict__ z, const float* __restrict__ dist_bins,
    const int*   __restrict__ pd_pairs,
    const float* __restrict__ u, const float* __restrict__ v,
    const __bf16* __restrict__ WbB, const float* __restrict__ bbias,
    const __bf16* __restrict__ WdB, const float* __restrict__ bdist,
    const float* __restrict__ g_ln, const float* __restrict__ b_ln,
    const __bf16* __restrict__ Wa1B, const float* __restrict__ ba1,
    const float* __restrict__ wa2, const float* __restrict__ ba2,
    const float* __restrict__ Wav, const float* __restrict__ bav,
    float* __restrict__ g_scores, float* __restrict__ g_vals)
{
    __shared__ __align__(16) __bf16 gate[16][776];   // 16 x 768 (+8 pad)
    __shared__ __align__(16) float  zpd [16][132];
    __shared__ __align__(16) __bf16 dbf [16][72];
    __shared__ __align__(16) float  hf  [16][260];
    __shared__ __align__(16) __bf16 hbf [16][264];
    __shared__ float red [16][17];
    __shared__ float red2[16][17];
    __shared__ float mu_s[16], rs_s[16];
    __shared__ float sc_part[8][16];
    __shared__ int   iidx[16], jidx[16];

    const int tid  = threadIdx.x;
    const int blk  = blockIdx.x;
    const int wave = tid >> 5;
    const int lane = tid & 31;
    const int l16  = lane & 15;
    const int g    = lane >> 4;

    if (tid < 16) {
        iidx[tid] = pd_pairs[(blk * 16 + tid) * 2 + 0];
        jidx[tid] = pd_pairs[(blk * 16 + tid) * 2 + 1];
    }
    __syncthreads();

#if HAS_ASYNC_LDS
    // gather z_pd tile via async global->LDS DMA (no VGPR round-trip, ASYNCcnt)
    for (int idx = tid; idx < 16 * (CP / 4); idx += 256) {   // 512 x b128
        int m = idx >> 5, n4 = idx & 31;
        long flat = (long)iidx[m] * LC + jidx[m];
        __builtin_amdgcn_global_load_async_to_lds_b128(
            (__attribute__((address_space(1))) v4i_t*)(z + flat * CP + n4 * 4),
            (__attribute__((address_space(3))) v4i_t*)&zpd[m][n4 * 4],
            0, 0);
    }
#endif

    // build gate = [u_i | v_j | u_i*v_j] (16 x 768) as bf16
    for (int idx = tid; idx < 16 * 768; idx += 256) {
        int m = idx / 768, k = idx - m * 768;
        float val;
        if (k < HID)            val = u[iidx[m] * HID + k];
        else if (k < 2 * HID)   val = v[jidx[m] * HID + (k - HID)];
        else { int c = k - 2 * HID; val = u[iidx[m] * HID + c] * v[jidx[m] * HID + c]; }
        gate[m][k] = (__bf16)val;
    }
#if !HAS_ASYNC_LDS
    // gather z_pd tile (f32) through VGPRs
    for (int idx = tid; idx < 16 * CP; idx += 256) {
        int m = idx >> 7, n = idx & (CP - 1);
        long flat = (long)iidx[m] * LC + jidx[m];
        zpd[m][n] = z[flat * CP + n];
    }
#endif
    // gather dist tile (needs f32->bf16 conversion, stays on VALU path)
    for (int idx = tid; idx < 16 * BB; idx += 256) {
        int m = idx >> 6, n = idx & (BB - 1);
        long flat = (long)iidx[m] * LC + jidx[m];
        dbf[m][n] = (__bf16)dist_bins[flat * BB + n];
    }
#if HAS_ASYNC_LDS
  #if __has_builtin(__builtin_amdgcn_s_wait_asynccnt)
    __builtin_amdgcn_s_wait_asynccnt(0);
  #else
    asm volatile("s_wait_asynccnt 0" ::: "memory");
  #endif
#endif
    __syncthreads();

    const int n0 = wave * 16 + l16;

    // ---- GEMM1: z_hat[m][n] = sum_k gate[m][k] * Wbias[n][k] + z_pd + bbias
    {
        v8f acc = {};
        const __bf16* bptr = WbB + n0 * (3 * HID) + g * 16;
        const __bf16* aptr = &gate[l16][g * 16];
        #pragma unroll 4
        for (int kb = 0; kb < 24; ++kb) {
            if (kb + 1 < 24) __builtin_prefetch(bptr + (kb + 1) * 32, 0, 3);
            AFrag A;
            A.h[0] = *reinterpret_cast<const v8bf*>(aptr + kb * 32);
            A.h[1] = *reinterpret_cast<const v8bf*>(aptr + kb * 32 + 8);
            v16bf B = *reinterpret_cast<const v16bf*>(bptr + kb * 32);
            acc = __builtin_amdgcn_wmma_f32_16x16x32_bf16(
                false, A.v, false, B, (short)0, acc, false, false);
        }
        float bb = bbias[n0];
        #pragma unroll
        for (int q = 0; q < 8; ++q) {
            int m = q + 8 * g;
            hf[m][n0] = acc[q] + zpd[m][n0] + bb;
        }
    }

    // ---- GEMM2: d_proj[m][n] = sum_k d[m][k] * Wdist[n][k] + bdist
    {
        v8f acc = {};
        const __bf16* bptr = WdB + n0 * BB + g * 16;
        const __bf16* aptr = &dbf[l16][g * 16];
        #pragma unroll
        for (int kb = 0; kb < 2; ++kb) {
            AFrag A;
            A.h[0] = *reinterpret_cast<const v8bf*>(aptr + kb * 32);
            A.h[1] = *reinterpret_cast<const v8bf*>(aptr + kb * 32 + 8);
            v16bf B = *reinterpret_cast<const v16bf*>(bptr + kb * 32);
            acc = __builtin_amdgcn_wmma_f32_16x16x32_bf16(
                false, A.v, false, B, (short)0, acc, false, false);
        }
        float bd = bdist[n0];
        #pragma unroll
        for (int q = 0; q < 8; ++q) {
            int m = q + 8 * g;
            hf[m][CP + n0] = acc[q] + bd;
        }
    }
    __syncthreads();

    // ---- LayerNorm over 256 channels
    {
        int r = tid >> 4, s = tid & 15;
        float sum = 0.f, sq = 0.f;
        #pragma unroll
        for (int c0 = 0; c0 < 16; ++c0) {
            float x = hf[r][s * 16 + c0];
            sum += x; sq += x * x;
        }
        red[r][s] = sum; red2[r][s] = sq;
    }
    __syncthreads();
    if (tid < 16) {
        float sum = 0.f, sq = 0.f;
        #pragma unroll
        for (int s = 0; s < 16; ++s) { sum += red[tid][s]; sq += red2[tid][s]; }
        float mu  = sum * (1.f / 256.f);
        float var = sq * (1.f / 256.f) - mu * mu;
        mu_s[tid] = mu;
        rs_s[tid] = rsqrtf(var + LN_EPS);
    }
    __syncthreads();
    for (int idx = tid; idx < 16 * 256; idx += 256) {
        int r = idx >> 8, c = idx & 255;
        float x = (hf[r][c] - mu_s[r]) * rs_s[r] * g_ln[c] + b_ln[c];
        hf[r][c]  = x;
        hbf[r][c] = (__bf16)x;
    }
    __syncthreads();

    // ---- GEMM3: t1 = tanh(h @ Wa1.T + ba1); score_part = sum_n t1*wa2[n]
    {
        float b1 = ba1[n0];
        v8f acc = { b1, b1, b1, b1, b1, b1, b1, b1 };
        const __bf16* bptr = Wa1B + n0 * (2 * CP) + g * 16;
        const __bf16* aptr = &hbf[l16][g * 16];
        #pragma unroll
        for (int kb = 0; kb < 8; ++kb) {
            AFrag A;
            A.h[0] = *reinterpret_cast<const v8bf*>(aptr + kb * 32);
            A.h[1] = *reinterpret_cast<const v8bf*>(aptr + kb * 32 + 8);
            v16bf B = *reinterpret_cast<const v16bf*>(bptr + kb * 32);
            acc = __builtin_amdgcn_wmma_f32_16x16x32_bf16(
                false, A.v, false, B, (short)0, acc, false, false);
        }
        float w2 = wa2[n0];
        #pragma unroll
        for (int q = 0; q < 8; ++q) {
            float sv = tanhf(acc[q]) * w2;
            #pragma unroll
            for (int off = 8; off > 0; off >>= 1)
                sv += __shfl_xor(sv, off, 16);
            if (l16 == 0) sc_part[wave][q + 8 * g] = sv;
        }
    }

    // ---- vals[m] = h[m] . Wav + bav (f32 dot)
    {
        int r = tid >> 4, s = tid & 15;
        float sum = 0.f;
        #pragma unroll
        for (int c0 = 0; c0 < 16; ++c0) {
            int c = s * 16 + c0;
            sum += hf[r][c] * Wav[c];
        }
        red[r][s] = sum;
    }
    __syncthreads();
    if (tid < 16) {
        float sc = ba2[0];
        #pragma unroll
        for (int w = 0; w < 8; ++w) sc += sc_part[w][tid];
        float vl = bav[0];
        #pragma unroll
        for (int s = 0; s < 16; ++s) vl += red[tid][s];
        g_scores[blk * 16 + tid] = sc;
        g_vals  [blk * 16 + tid] = vl;
    }
}

// ---------------------------------------------------------------- softmax + pooling (deterministic)
__global__ __launch_bounds__(256) void k_pmax(const float* __restrict__ s, float* __restrict__ pmax)
{
    __shared__ float red[256];
    int t = threadIdx.x, i = blockIdx.x * 256 + t;
    red[t] = (i < EP) ? s[i] : -INFINITY;
    __syncthreads();
    for (int off = 128; off > 0; off >>= 1) {
        if (t < off) red[t] = fmaxf(red[t], red[t + off]);
        __syncthreads();
    }
    if (t == 0) pmax[blockIdx.x] = red[0];
}

__global__ __launch_bounds__(256) void k_rmax(const float* __restrict__ pmax, int n, float* __restrict__ gmax)
{
    __shared__ float red[256];
    int t = threadIdx.x;
    float v = -INFINITY;
    for (int i = t; i < n; i += 256) v = fmaxf(v, pmax[i]);
    red[t] = v;
    __syncthreads();
    for (int off = 128; off > 0; off >>= 1) {
        if (t < off) red[t] = fmaxf(red[t], red[t + off]);
        __syncthreads();
    }
    if (t == 0) gmax[0] = red[0];
}

__global__ __launch_bounds__(256) void k_exp(const float* __restrict__ s, const float* __restrict__ vals,
                                             const float* __restrict__ gmax,
                                             float* __restrict__ ew, float* __restrict__ psum, float* __restrict__ pwv)
{
    __shared__ float r1[256], r2[256];
    int t = threadIdx.x, i = blockIdx.x * 256 + t;
    float e = 0.f, wv = 0.f;
    if (i < EP) {
        e = __expf((s[i] - gmax[0]) * INV_TEMP);
        ew[i] = e;
        wv = e * vals[i];
    }
    r1[t] = e; r2[t] = wv;
    __syncthreads();
    for (int off = 128; off > 0; off >>= 1) {
        if (t < off) { r1[t] += r1[t + off]; r2[t] += r2[t + off]; }
        __syncthreads();
    }
    if (t == 0) { psum[blockIdx.x] = r1[0]; pwv[blockIdx.x] = r2[0]; }
}

__global__ __launch_bounds__(256) void k_rsum(const float* __restrict__ psum, const float* __restrict__ pwv,
                                              int n, float* __restrict__ out0, float* __restrict__ invS)
{
    __shared__ float r1[256], r2[256];
    int t = threadIdx.x;
    float s = 0.f, wv = 0.f;
    for (int i = t; i < n; i += 256) { s += psum[i]; wv += pwv[i]; }
    r1[t] = s; r2[t] = wv;
    __syncthreads();
    for (int off = 128; off > 0; off >>= 1) {
        if (t < off) { r1[t] += r1[t + off]; r2[t] += r2[t + off]; }
        __syncthreads();
    }
    if (t == 0) {
        out0[0] = r2[0] / r1[0];   // pooled = sum(w*vals) = sum(e*vals)/sum(e)
        invS[0] = 1.f / r1[0];
    }
}

__global__ __launch_bounds__(256) void k_normw(const float* __restrict__ ew, const float* __restrict__ invS,
                                               float* __restrict__ wout)
{
    int i = blockIdx.x * 256 + threadIdx.x;
    if (i < EP) wout[i] = ew[i] * invS[0];
}

// ---------------------------------------------------------------- launcher
extern "C" void kernel_launch(void* const* d_in, const int* in_sizes, int n_in,
                              void* d_out, int out_size, void* d_ws, size_t ws_size,
                              hipStream_t stream)
{
    const float* z         = (const float*)d_in[0];
    const float* s_proxy   = (const float*)d_in[1];
    const float* dist_bins = (const float*)d_in[2];
    const int*   pd_pairs  = (const int*)  d_in[3];
    const float* Wpu  = (const float*)d_in[4];
    const float* bpu  = (const float*)d_in[5];
    const float* Wpv  = (const float*)d_in[6];
    const float* bpv  = (const float*)d_in[7];
    const float* Wbias= (const float*)d_in[8];
    const float* bbias= (const float*)d_in[9];
    const float* Wdist= (const float*)d_in[10];
    const float* bdist= (const float*)d_in[11];
    const float* g_ln = (const float*)d_in[12];
    const float* b_ln = (const float*)d_in[13];
    const float* Wa1  = (const float*)d_in[14];
    const float* ba1  = (const float*)d_in[15];
    const float* wa2  = (const float*)d_in[16];
    const float* ba2  = (const float*)d_in[17];
    const float* Wav  = (const float*)d_in[18];
    const float* bav  = (const float*)d_in[19];

    // workspace carve-out (256B aligned slices)
    size_t off = 0;
    auto alloc = [&](size_t bytes) -> char* {
        char* p = (char*)d_ws + off;
        off += (bytes + 255) & ~(size_t)255;
        return p;
    };
    float*  u_buf    = (float*) alloc((size_t)LC * HID * 4);
    float*  v_buf    = (float*) alloc((size_t)LC * HID * 4);
    __bf16* WbB      = (__bf16*)alloc((size_t)CP * 3 * HID * 2);
    __bf16* WdB      = (__bf16*)alloc((size_t)CP * BB * 2);
    __bf16* Wa1B     = (__bf16*)alloc((size_t)AH * 2 * CP * 2);
    float*  g_scores = (float*) alloc((size_t)EP * 4);
    float*  g_vals   = (float*) alloc((size_t)EP * 4);
    float*  ew       = (float*) alloc((size_t)EP * 4);
    const int NB = (EP + 255) / 256;   // 1172
    float* pmax = (float*)alloc((size_t)NB * 4);
    float* psum = (float*)alloc((size_t)NB * 4);
    float* pwv  = (float*)alloc((size_t)NB * 4);
    float* gmax = (float*)alloc(4);
    float* invS = (float*)alloc(4);

    float* out = (float*)d_out;   // out[0]=pooled, out[1..EP]=w

    k_proj_uv<<<LC, 256, 0, stream>>>(s_proxy, Wpu, bpu, Wpv, bpv, u_buf, v_buf);
    k_cvt<<<(CP * 3 * HID + 255) / 256, 256, 0, stream>>>(Wbias, Wdist, Wa1, WbB, WdB, Wa1B);
    k_pairs<<<EP / 16, 256, 0, stream>>>(z, dist_bins, pd_pairs, u_buf, v_buf,
                                         WbB, bbias, WdB, bdist, g_ln, b_ln,
                                         Wa1B, ba1, wa2, ba2, Wav, bav,
                                         g_scores, g_vals);
    k_pmax<<<NB, 256, 0, stream>>>(g_scores, pmax);
    k_rmax<<<1, 256, 0, stream>>>(pmax, NB, gmax);
    k_exp<<<NB, 256, 0, stream>>>(g_scores, g_vals, gmax, ew, psum, pwv);
    k_rsum<<<1, 256, 0, stream>>>(psum, pwv, NB, out, invS);
    k_normw<<<NB, 256, 0, stream>>>(ew, invS, out + 1);
}
